// BatchTopKTranscoder_2654289789134
// MI455X (gfx1250) — compile-verified, matrix operand order checked by reading, not run on hardware
//
#include <hip/hip_runtime.h>
#include <hip/hip_bf16.h>

#define BATCH 2048
#define D_IN 768
#define D_OUT 768
#define DICT 16384
#define TOPK_TOTAL (32 * 2048)
#define TOPK_AUX 512
#define NDEAD_TH 15360
#define MAX_ROW_NNZ 512

// ctl indices
#define CTL_B1 0
#define CTL_CGT 1
#define CTL_TSTRICT 2
#define CTL_TTIE 3
#define CTL_REMAIN 4
#define CTL_TIECTR 5
#define CTL_L0 6
#define CTL_DEADANY 7
#define CTL_NUMDEAD 8

// accum indices (doubles)
#define ACC_L1 0
#define ACC_L2 1
#define ACC_AUX 2

typedef __bf16 v16bf __attribute__((ext_vector_type(16)));
typedef float v8f __attribute__((ext_vector_type(8)));

// TDM descriptor operand types (probe-verified arity; see CDNA5 ISA)
typedef unsigned tdm_v4u __attribute__((ext_vector_type(4)));
typedef int tdm_v8i __attribute__((ext_vector_type(8)));
typedef int tdm_v4i __attribute__((ext_vector_type(4)));

#if defined(__AMDGCN__) && defined(__has_builtin)
#if __has_builtin(__builtin_amdgcn_tensor_load_to_lds)
#define USE_TDM 1
#endif
#endif
#ifndef USE_TDM
#define USE_TDM 0
#endif

union Frag { v16bf v; unsigned u[8]; };

__device__ inline unsigned short f2bf(float f) {
  unsigned u = __float_as_uint(f);
  unsigned r = u + 0x7FFFu + ((u >> 16) & 1u);   // RNE
  return (unsigned short)(r >> 16);
}

__device__ inline float blockReduceSum(float v, float* sbuf) {
  const int t = threadIdx.x;
  sbuf[t] = v;
  __syncthreads();
  for (int s = 128; s > 0; s >>= 1) {
    if (t < s) sbuf[t] += sbuf[t + s];
    __syncthreads();
  }
  float r = sbuf[0];
  __syncthreads();
  return r;
}

// ---------------------------------------------------------------------------
// K1: per-row standardize x (-> bf16 x_enc = x_proc - b_dec) and y (-> y_proc)
// blocks [0,2048): x rows; [2048,4096): y rows. 256 thr, 3 elems/thr.
// ---------------------------------------------------------------------------
__global__ __launch_bounds__(256) void k_rowstats(
    const float* __restrict__ x_in, const float* __restrict__ y_tgt,
    const float* __restrict__ b_dec, unsigned short* __restrict__ A_bf,
    float* __restrict__ y_proc, float* __restrict__ y_mean, float* __restrict__ y_std) {
  __shared__ float sbuf[256];
  const int row = blockIdx.x & (BATCH - 1);
  const bool isY = blockIdx.x >= BATCH;
  const float* p = (isY ? y_tgt : x_in) + (size_t)row * D_IN;
  float v[3], s = 0.f, sq = 0.f;
#pragma unroll
  for (int j = 0; j < 3; ++j) {
    v[j] = p[threadIdx.x + j * 256];
    s += v[j];
    sq += v[j] * v[j];
  }
  s = blockReduceSum(s, sbuf);
  sq = blockReduceSum(sq, sbuf);
  const float mean = s * (1.0f / D_IN);
  const float var = (sq - s * mean) * (1.0f / (D_IN - 1));  // ddof=1
  const float sd = sqrtf(fmaxf(var, 0.0f));
  const float inv = 1.0f / (sd + 1e-5f);
  if (isY) {
    if (threadIdx.x == 0) { y_mean[row] = mean; y_std[row] = sd; }
#pragma unroll
    for (int j = 0; j < 3; ++j) {
      const int d = threadIdx.x + j * 256;
      y_proc[(size_t)row * D_OUT + d] = (v[j] - mean) * inv;
    }
  } else {
#pragma unroll
    for (int j = 0; j < 3; ++j) {
      const int d = threadIdx.x + j * 256;
      A_bf[(size_t)row * D_IN + d] = f2bf((v[j] - mean) * inv - b_dec[d]);
    }
  }
}

// ---------------------------------------------------------------------------
// K2: W_enc f32 -> bf16
// ---------------------------------------------------------------------------
__global__ void k_cvt_bf16(const float* __restrict__ src, unsigned short* __restrict__ dst, size_t n) {
  for (size_t i = (size_t)blockIdx.x * blockDim.x + threadIdx.x; i < n;
       i += (size_t)gridDim.x * blockDim.x)
    dst[i] = f2bf(src[i]);
}

// ---------------------------------------------------------------------------
// K3: encoder GEMM: acts_dense[2048,16384] = relu(A[2048,768] @ W[768,16384] + b_enc)
// bf16 WMMA 16x16x32, block tile 128x128, BK=32, 8 waves (2x4), wave tile 64x32.
// A staged manually as packed k-pairs (pitch 17, conflict-free).
// B staged via Tensor Data Mover: wave 0 issues a 2D D# (tile 32x128 bf16 out of
// the [768,16384] tensor, dim0_stride=16384) into LDS, waits TENSORcnt, then the
// workgroup barrier releases consumers. Fallback: manual transpose staging.
// ISA 16-bit A/B lane layout: m/n = lane&15, half = lane>>4, vgpr v -> k2 =
// (v&3) + 4*half + 8*(v>>2)   (K pair index, K = 2*k2).
// ---------------------------------------------------------------------------
__global__ __launch_bounds__(256) void k_encoder(
    const unsigned short* __restrict__ Abf, const unsigned short* __restrict__ Wbf,
    const float* __restrict__ b_enc, float* __restrict__ acts_dense) {
  __shared__ unsigned As[128][17];
#if USE_TDM
  __shared__ unsigned short Bs2[32][128];  // row-major [k][n], filled by TDM
#else
  __shared__ unsigned Bs[128][17];         // pair-packed [n][k2]
#endif
  const int t = threadIdx.x;
  const int lane = t & 31;
  const int w = t >> 5;
  const int wm = w & 1;    // wave row (2)
  const int wn = w >> 1;   // wave col (4)
  const int mrow = lane & 15;
  const int half = lane >> 4;
  const int blockM = blockIdx.y * 128;
  const int blockN = blockIdx.x * 128;

  v8f acc[4][2];
#pragma unroll
  for (int mi = 0; mi < 4; ++mi)
#pragma unroll
    for (int ni = 0; ni < 2; ++ni) {
      v8f z = {0.f, 0.f, 0.f, 0.f, 0.f, 0.f, 0.f, 0.f};
      acc[mi][ni] = z;
    }

  // A-tile loader: 128 rows x 16 uints; thread -> (row, 8-uint half-row)
  const int lam = t >> 1;
  const int lac = (t & 1) * 8;
#if !USE_TDM
  // B-tile loader: thread -> (k-pair, 8 consecutive n)
  const int lbk2 = t & 15;
  const int lbn0 = (t >> 4) * 8;
#endif

  for (int k0 = 0; k0 < D_IN; k0 += 32) {
    { // stage A: row-major k-pairs, directly packed
      const unsigned* g = (const unsigned*)(Abf + (size_t)(blockM + lam) * D_IN + k0) + lac;
      if (k0 + 32 < D_IN) __builtin_prefetch((const void*)(g + 16), 0, 1);
#pragma unroll
      for (int j = 0; j < 8; ++j) As[lam][lac + j] = g[j];
    }
#if USE_TDM
    if (w == 0) {  // TDM is a per-wave op (EXEC ignored): issue from one wave only
      const unsigned lds_off = (unsigned)(unsigned long long)&Bs2[0][0];
      const unsigned long long ga =
          (unsigned long long)(const unsigned short*)(Wbf + (size_t)k0 * DICT + blockN);
      // D# group 0: count=1, lds_addr, global_addr[56:0], type=2
      tdm_v4u g0 = {1u, lds_off, (unsigned)ga,
                    (unsigned)((ga >> 32) & 0x01FFFFFFull) | (2u << 30)};
      // D# group 1: data_size=2B; tensor_dim0=16384, tensor_dim1=768;
      // tile_dim0=128, tile_dim1=32 (2D); tensor_dim0_stride=16384.
      tdm_v8i g1 = {(int)(1u << 16),
                    (int)(((unsigned)DICT & 0xFFFFu) << 16),
                    (int)((((unsigned)D_IN & 0xFFFFu) << 16) | ((unsigned)DICT >> 16)),
                    (int)((128u << 16) | ((unsigned)D_IN >> 16)),
                    32,
                    (int)DICT,
                    0, 0};
      tdm_v4i gz = {0, 0, 0, 0};
#if defined(__clang_major__) && __clang_major__ >= 23
      tdm_v8i gz8 = {0, 0, 0, 0, 0, 0, 0, 0};
      __builtin_amdgcn_tensor_load_to_lds(g0, g1, gz, gz, gz8, 0);
#else
      __builtin_amdgcn_tensor_load_to_lds(g0, g1, gz, gz, 0);
#endif
      __builtin_amdgcn_s_wait_tensorcnt(0);
    }
#else
    { // stage B: transpose [k][n] -> packed [n][k2] (pair k,k+1 in one uint)
      const unsigned short* r0 = Wbf + (size_t)(k0 + 2 * lbk2) * DICT + blockN + lbn0;
      const unsigned short* r1 = r0 + DICT;
      if (k0 + 32 < D_IN) __builtin_prefetch((const void*)(r0 + (size_t)32 * DICT), 0, 1);
      const uint4 A0 = *(const uint4*)r0;
      const uint4 B0 = *(const uint4*)r1;
      const unsigned ra[4] = {A0.x, A0.y, A0.z, A0.w};
      const unsigned rb[4] = {B0.x, B0.y, B0.z, B0.w};
#pragma unroll
      for (int j = 0; j < 4; ++j) {
        Bs[lbn0 + 2 * j + 0][lbk2] = (ra[j] & 0xFFFFu) | ((rb[j] & 0xFFFFu) << 16);
        Bs[lbn0 + 2 * j + 1][lbk2] = (ra[j] >> 16) | (rb[j] & 0xFFFF0000u);
      }
    }
#endif
    __syncthreads();

    Frag af[4], bfr[2];
#pragma unroll
    for (int v8i = 0; v8i < 8; ++v8i) {
      const int k2 = (v8i & 3) + 4 * half + 8 * (v8i >> 2);
#pragma unroll
      for (int mi = 0; mi < 4; ++mi) af[mi].u[v8i] = As[wm * 64 + mi * 16 + mrow][k2];
#if USE_TDM
      const int kk = 2 * k2;
#pragma unroll
      for (int ni = 0; ni < 2; ++ni) {
        const int n = wn * 32 + ni * 16 + mrow;
        bfr[ni].u[v8i] = (unsigned)Bs2[kk][n] | ((unsigned)Bs2[kk + 1][n] << 16);
      }
#else
#pragma unroll
      for (int ni = 0; ni < 2; ++ni) bfr[ni].u[v8i] = Bs[wn * 32 + ni * 16 + mrow][k2];
#endif
    }
#pragma unroll
    for (int mi = 0; mi < 4; ++mi)
#pragma unroll
      for (int ni = 0; ni < 2; ++ni)
        acc[mi][ni] = __builtin_amdgcn_wmma_f32_16x16x32_bf16(
            false, af[mi].v, false, bfr[ni].v, (short)0, acc[mi][ni], false, false);
    __syncthreads();
  }

  // epilogue: C/D layout: vgpr r -> M = r + 8*half; N = lane&15
#pragma unroll
  for (int mi = 0; mi < 4; ++mi)
#pragma unroll
    for (int ni = 0; ni < 2; ++ni) {
      const int col = blockN + wn * 32 + ni * 16 + mrow;
      const float be = b_enc[col];
#pragma unroll
      for (int r = 0; r < 8; ++r) {
        const int row = blockM + wm * 64 + mi * 16 + half * 8 + r;
        acts_dense[(size_t)row * DICT + col] = fmaxf(acc[mi][ni][r] + be, 0.0f);
      }
    }
}

// ---------------------------------------------------------------------------
// K5/K7: global histograms over positive-float bits (monotonic for v>0)
// level 1: bits>>19 (4096 bins); level 2: bits[18:7] within the level-1 bin.
// ---------------------------------------------------------------------------
__global__ __launch_bounds__(256) void k_hist1(const float* __restrict__ acts, unsigned* __restrict__ hist) {
  __shared__ unsigned h[4096];
  for (int i = threadIdx.x; i < 4096; i += 256) h[i] = 0;
  __syncthreads();
  const size_t n = (size_t)BATCH * DICT;
  for (size_t i = (size_t)blockIdx.x * blockDim.x + threadIdx.x; i < n;
       i += (size_t)gridDim.x * blockDim.x) {
    const float v = acts[i];
    if (v > 0.0f) atomicAdd(&h[__float_as_uint(v) >> 19], 1u);
  }
  __syncthreads();
  for (int i = threadIdx.x; i < 4096; i += 256)
    if (h[i]) atomicAdd(&hist[i], h[i]);
}

__global__ void k_thresh1(const unsigned* __restrict__ hist, unsigned* __restrict__ ctl) {
  unsigned acc = 0;
  int b = -1;
  for (int i = 4095; i >= 0; --i) {
    const unsigned c = hist[i];
    if (acc + c >= (unsigned)TOPK_TOTAL) { b = i; break; }
    acc += c;
  }
  if (b < 0) {  // fewer positives than K: keep all positives
    ctl[CTL_B1] = 0xFFFFFFFFu;
    ctl[CTL_CGT] = acc;
    ctl[CTL_TSTRICT] = 1u;
    ctl[CTL_TTIE] = 1u;
    ctl[CTL_REMAIN] = 0u;
  } else {
    ctl[CTL_B1] = (unsigned)b;
    ctl[CTL_CGT] = acc;
  }
}

__global__ __launch_bounds__(256) void k_hist2(const float* __restrict__ acts,
                                               const unsigned* __restrict__ ctl,
                                               unsigned* __restrict__ hist) {
  const unsigned b1 = ctl[CTL_B1];
  if (b1 == 0xFFFFFFFFu) return;
  __shared__ unsigned h[4096];
  for (int i = threadIdx.x; i < 4096; i += 256) h[i] = 0;
  __syncthreads();
  const size_t n = (size_t)BATCH * DICT;
  for (size_t i = (size_t)blockIdx.x * blockDim.x + threadIdx.x; i < n;
       i += (size_t)gridDim.x * blockDim.x) {
    const float v = acts[i];
    if (v > 0.0f) {
      const unsigned bits = __float_as_uint(v);
      if ((bits >> 19) == b1) atomicAdd(&h[(bits >> 7) & 0xFFFu], 1u);
    }
  }
  __syncthreads();
  for (int i = threadIdx.x; i < 4096; i += 256)
    if (h[i]) atomicAdd(&hist[i], h[i]);
}

__global__ void k_thresh2(const unsigned* __restrict__ hist, unsigned* __restrict__ ctl) {
  const unsigned b1 = ctl[CTL_B1];
  if (b1 == 0xFFFFFFFFu) return;
  const unsigned Kp = (unsigned)TOPK_TOTAL - ctl[CTL_CGT];
  unsigned acc = 0;
  int s = -1;
  for (int i = 4095; i >= 0; --i) {
    const unsigned c = hist[i];
    if (acc + c >= Kp) { s = i; break; }
    acc += c;
  }
  if (s < 0) {  // should not happen; keep whole bin
    ctl[CTL_TTIE] = b1 << 19;
    ctl[CTL_TSTRICT] = b1 << 19;
    ctl[CTL_REMAIN] = 0u;
    return;
  }
  ctl[CTL_TTIE] = (b1 << 19) | ((unsigned)s << 7);
  ctl[CTL_TSTRICT] = (b1 << 19) + (((unsigned)s + 1u) << 7);  // s=4095 rolls into next bin
  ctl[CTL_REMAIN] = Kp - acc;
}

// ---------------------------------------------------------------------------
// K9: scatter: produce dense `acts` output, per-row nnz lists, col_active,
// l0 count, l1 sum.
// ---------------------------------------------------------------------------
__global__ __launch_bounds__(256) void k_scatter(
    const float* __restrict__ acts_dense, unsigned* __restrict__ ctl,
    float* __restrict__ acts_out, unsigned* __restrict__ row_nnz,
    unsigned* __restrict__ row_cols, float* __restrict__ row_vals,
    unsigned* __restrict__ col_active, double* __restrict__ accum) {
  const unsigned Ts = ctl[CTL_TSTRICT];
  const unsigned Tt = ctl[CTL_TTIE];
  const unsigned rem = ctl[CTL_REMAIN];
  const size_t n = (size_t)BATCH * DICT;
  for (size_t i = (size_t)blockIdx.x * blockDim.x + threadIdx.x; i < n;
       i += (size_t)gridDim.x * blockDim.x) {
    const float v = acts_dense[i];
    float out = 0.0f;
    if (v > 0.0f) {
      const unsigned bits = __float_as_uint(v);
      bool keep = false;
      if (bits >= Ts) keep = true;
      else if (bits >= Tt && atomicAdd(&ctl[CTL_TIECTR], 1u) < rem) keep = true;
      if (keep) {
        out = v;
        const unsigned row = (unsigned)(i >> 14);
        const unsigned col = (unsigned)(i & (DICT - 1));
        col_active[col] = 1u;
        const unsigned slot = atomicAdd(&row_nnz[row], 1u);
        if (slot < MAX_ROW_NNZ) {
          row_cols[(size_t)row * MAX_ROW_NNZ + slot] = col;
          row_vals[(size_t)row * MAX_ROW_NNZ + slot] = v;
        }
        atomicAdd(&accum[ACC_L1], (double)v);
        atomicAdd(&ctl[CTL_L0], 1u);
      }
    }
    acts_out[i] = out;
  }
}

// ---------------------------------------------------------------------------
// K10: dead-feature bookkeeping
// ---------------------------------------------------------------------------
__global__ __launch_bounds__(256) void k_dead(
    const unsigned* __restrict__ col_active, const int* __restrict__ nba_in,
    unsigned* __restrict__ dead, float* __restrict__ nba_out, unsigned* __restrict__ ctl) {
  const int c = blockIdx.x * 256 + threadIdx.x;
  if (c >= DICT) return;
  const int nba = col_active[c] ? 0 : (nba_in[c] + 1);
  nba_out[c] = (float)nba;
  const unsigned d = (nba >= NDEAD_TH) ? 1u : 0u;
  dead[c] = d;
  if (d) atomicOr(&ctl[CTL_DEADANY], 1u);
  if (nba > NDEAD_TH) atomicAdd(&ctl[CTL_NUMDEAD], 1u);
}

// ---------------------------------------------------------------------------
// K11: sparse decoder: y_pred = acts @ W_dec + b_dec  (W_dec rows from L2),
// y_pred_out = y_pred * y_std + y_mean, l2 partial sum.
// one block per row; 256 thr * 3 outputs.
// ---------------------------------------------------------------------------
__global__ __launch_bounds__(256) void k_decode(
    const unsigned* __restrict__ row_nnz, const unsigned* __restrict__ row_cols,
    const float* __restrict__ row_vals, const float* __restrict__ W_dec,
    const float* __restrict__ b_dec, const float* __restrict__ y_proc,
    const float* __restrict__ y_mean, const float* __restrict__ y_std,
    float* __restrict__ y_pred, float* __restrict__ y_out, double* __restrict__ accum) {
  __shared__ float sbuf[256];
  const int row = blockIdx.x;
  const int t = threadIdx.x;
  unsigned nnz = row_nnz[row];
  if (nnz > MAX_ROW_NNZ) nnz = MAX_ROW_NNZ;
  float a0 = b_dec[t], a1 = b_dec[t + 256], a2 = b_dec[t + 512];
  const size_t lb = (size_t)row * MAX_ROW_NNZ;
  for (unsigned j = 0; j < nnz; ++j) {
    const unsigned col = row_cols[lb + j];
    const float val = row_vals[lb + j];
    const float* wr = W_dec + (size_t)col * D_OUT;
    a0 += val * wr[t];
    a1 += val * wr[t + 256];
    a2 += val * wr[t + 512];
  }
  const size_t base = (size_t)row * D_OUT;
  y_pred[base + t] = a0;
  y_pred[base + t + 256] = a1;
  y_pred[base + t + 512] = a2;
  const float m = y_mean[row], s = y_std[row];
  y_out[base + t] = a0 * s + m;
  y_out[base + t + 256] = a1 * s + m;
  y_out[base + t + 512] = a2 * s + m;
  const float e0 = a0 - y_proc[base + t];
  const float e1 = a1 - y_proc[base + t + 256];
  const float e2 = a2 - y_proc[base + t + 512];
  const float sq = blockReduceSum(e0 * e0 + e1 * e1 + e2 * e2, sbuf);
  if (t == 0) atomicAdd(&accum[ACC_L2], (double)sq);
}

// ---------------------------------------------------------------------------
// K12: per-row top-512 over positive acts on dead columns (zeros never matter
// for the aux decode). 256-bin exponent histogram + tie cap.
// ---------------------------------------------------------------------------
__global__ __launch_bounds__(256) void k_auxsel(
    const float* __restrict__ acts_dense, const unsigned* __restrict__ dead,
    unsigned* __restrict__ aux_nnz, unsigned* __restrict__ aux_cols,
    float* __restrict__ aux_vals) {
  __shared__ unsigned h[256];
  __shared__ unsigned sctl[4];  // bstar, remain, tie ctr, slot ctr
  const int row = blockIdx.x;
  const int t = threadIdx.x;
  h[t] = 0;
  if (t < 4) sctl[t] = 0;
  __syncthreads();
  const float* ar = acts_dense + (size_t)row * DICT;
  for (int c = t; c < DICT; c += 256) {
    if (dead[c]) {
      const float v = ar[c];
      if (v > 0.0f) atomicAdd(&h[__float_as_uint(v) >> 23], 1u);
    }
  }
  __syncthreads();
  if (t == 0) {
    unsigned tot = 0;
    for (int i = 0; i < 256; ++i) tot += h[i];
    if (tot <= (unsigned)TOPK_AUX) {
      sctl[0] = 0;
      sctl[1] = 0xFFFFFFFFu;  // keep-all sentinel
    } else {
      unsigned acc = 0;
      int b = 0;
      for (int i = 255; i >= 0; --i) {
        if (acc + h[i] >= (unsigned)TOPK_AUX) { b = i; break; }
        acc += h[i];
      }
      sctl[0] = (unsigned)b;
      sctl[1] = (unsigned)TOPK_AUX - acc;
    }
  }
  __syncthreads();
  const unsigned bstar = sctl[0], rem = sctl[1];
  for (int c = t; c < DICT; c += 256) {
    if (dead[c]) {
      const float v = ar[c];
      if (v > 0.0f) {
        bool keep = false;
        if (rem == 0xFFFFFFFFu) keep = true;
        else {
          const unsigned bin = __float_as_uint(v) >> 23;
          if (bin > bstar) keep = true;
          else if (bin == bstar && atomicAdd(&sctl[2], 1u) < rem) keep = true;
        }
        if (keep) {
          const unsigned slot = atomicAdd(&sctl[3], 1u);
          if (slot < (unsigned)TOPK_AUX) {
            aux_cols[(size_t)row * TOPK_AUX + slot] = (unsigned)c;
            aux_vals[(size_t)row * TOPK_AUX + slot] = v;
          }
        }
      }
    }
  }
  __syncthreads();
  if (t == 0) aux_nnz[row] = (sctl[3] < (unsigned)TOPK_AUX) ? sctl[3] : (unsigned)TOPK_AUX;
}

// ---------------------------------------------------------------------------
// K13: aux decode + aux squared-error sum: sum((acts_aux@W_dec - (y_proc - y_pred))^2)
// ---------------------------------------------------------------------------
__global__ __launch_bounds__(256) void k_auxdec(
    const unsigned* __restrict__ aux_nnz, const unsigned* __restrict__ aux_cols,
    const float* __restrict__ aux_vals, const float* __restrict__ W_dec,
    const float* __restrict__ y_proc, const float* __restrict__ y_pred,
    double* __restrict__ accum) {
  __shared__ float sbuf[256];
  const int row = blockIdx.x;
  const int t = threadIdx.x;
  const unsigned nnz = aux_nnz[row];
  float a0 = 0.f, a1 = 0.f, a2 = 0.f;
  const size_t lb = (size_t)row * TOPK_AUX;
  for (unsigned j = 0; j < nnz; ++j) {
    const unsigned col = aux_cols[lb + j];
    const float val = aux_vals[lb + j];
    const float* wr = W_dec + (size_t)col * D_OUT;
    a0 += val * wr[t];
    a1 += val * wr[t + 256];
    a2 += val * wr[t + 512];
  }
  const size_t base = (size_t)row * D_OUT;
  const float e0 = a0 - (y_proc[base + t] - y_pred[base + t]);
  const float e1 = a1 - (y_proc[base + t + 256] - y_pred[base + t + 256]);
  const float e2 = a2 - (y_proc[base + t + 512] - y_pred[base + t + 512]);
  const float sq = blockReduceSum(e0 * e0 + e1 * e1 + e2 * e2, sbuf);
  if (t == 0) atomicAdd(&accum[ACC_AUX], (double)sq);
}

// ---------------------------------------------------------------------------
// K14: finalize scalar outputs
// order: loss, l2_loss, l0_norm, l1_norm, l1_loss, aux_loss, num_dead_features
// ---------------------------------------------------------------------------
__global__ void k_final(const double* __restrict__ accum, const unsigned* __restrict__ ctl,
                        float* __restrict__ outs) {
  const double denom = (double)BATCH * (double)D_OUT;
  const double l2 = accum[ACC_L2] / denom;
  const double l1n = accum[ACC_L1] / (double)BATCH;
  const double l1l = 0.0003 * l1n;
  const double l0 = (double)ctl[CTL_L0] / (double)BATCH;
  const double aux = ctl[CTL_DEADANY] ? (0.03125 * accum[ACC_AUX] / denom) : 0.0;
  const double loss = l2 + l1l + aux;
  outs[0] = (float)loss;
  outs[1] = (float)l2;
  outs[2] = (float)l0;
  outs[3] = (float)l1n;
  outs[4] = (float)l1l;
  outs[5] = (float)aux;
  outs[6] = (float)ctl[CTL_NUMDEAD];
}

// ---------------------------------------------------------------------------
extern "C" void kernel_launch(void* const* d_in, const int* in_sizes, int n_in,
                              void* d_out, int out_size, void* d_ws, size_t ws_size,
                              hipStream_t stream) {
  (void)in_sizes; (void)n_in; (void)out_size; (void)ws_size;
  const float* x_in = (const float*)d_in[0];
  const float* y_tgt = (const float*)d_in[1];
  const float* W_enc = (const float*)d_in[2];
  const float* b_enc = (const float*)d_in[3];
  const float* W_dec = (const float*)d_in[4];
  const float* b_dec = (const float*)d_in[5];
  const int* nba = (const int*)d_in[6];
  float* out = (float*)d_out;

  char* ws = (char*)d_ws;
  size_t off = 0;
  auto take = [&](size_t bytes) {
    size_t o = off;
    off += (bytes + 255) & ~(size_t)255;
    return o;
  };

  // ---- zeroed-every-call region (one contiguous memset) ----
  const size_t zero_base = off;
  unsigned* ctl = (unsigned*)(ws + take(64 * 4));
  double* accum = (double*)(ws + take(8 * 8));
  unsigned* hist1 = (unsigned*)(ws + take(4096 * 4));
  unsigned* hist2 = (unsigned*)(ws + take(4096 * 4));
  unsigned* col_active = (unsigned*)(ws + take((size_t)DICT * 4));
  unsigned* row_nnz = (unsigned*)(ws + take((size_t)BATCH * 4));
  unsigned* aux_nnz = (unsigned*)(ws + take((size_t)BATCH * 4));
  const size_t zero_bytes = off - zero_base;

  // ---- plain scratch ----
  unsigned short* A_bf = (unsigned short*)(ws + take((size_t)BATCH * D_IN * 2));
  unsigned short* W_bf = (unsigned short*)(ws + take((size_t)D_IN * DICT * 2));
  float* y_proc = (float*)(ws + take((size_t)BATCH * D_OUT * 4));
  float* y_mean = (float*)(ws + take((size_t)BATCH * 4));
  float* y_std = (float*)(ws + take((size_t)BATCH * 4));
  float* acts_dense = (float*)(ws + take((size_t)BATCH * DICT * 4));
  float* y_pred = (float*)(ws + take((size_t)BATCH * D_OUT * 4));
  unsigned* dead = (unsigned*)(ws + take((size_t)DICT * 4));
  unsigned* row_cols = (unsigned*)(ws + take((size_t)BATCH * MAX_ROW_NNZ * 4));
  float* row_vals = (float*)(ws + take((size_t)BATCH * MAX_ROW_NNZ * 4));
  unsigned* aux_cols = (unsigned*)(ws + take((size_t)BATCH * TOPK_AUX * 4));
  float* aux_vals = (float*)(ws + take((size_t)BATCH * TOPK_AUX * 4));

  // ---- output layout (tuple concat, all as f32) ----
  float* y_out_d = out;                                   // [B, D_OUT]
  float* acts_d = out + (size_t)BATCH * D_OUT;            // [B, DICT]
  float* scalars_d = acts_d + (size_t)BATCH * DICT;       // 7 scalars
  float* nba_d = scalars_d + 7;                           // [DICT]

  hipMemsetAsync(ws + zero_base, 0, zero_bytes, stream);

  k_rowstats<<<2 * BATCH, 256, 0, stream>>>(x_in, y_tgt, b_dec, A_bf, y_proc, y_mean, y_std);
  k_cvt_bf16<<<4096, 256, 0, stream>>>(W_enc, W_bf, (size_t)D_IN * DICT);

  dim3 ge(DICT / 128, BATCH / 128);
  k_encoder<<<ge, 256, 0, stream>>>(A_bf, W_bf, b_enc, acts_dense);

  k_hist1<<<2048, 256, 0, stream>>>(acts_dense, hist1);
  k_thresh1<<<1, 1, 0, stream>>>(hist1, ctl);
  k_hist2<<<2048, 256, 0, stream>>>(acts_dense, ctl, hist2);
  k_thresh2<<<1, 1, 0, stream>>>(hist2, ctl);
  k_scatter<<<2048, 256, 0, stream>>>(acts_dense, ctl, acts_d, row_nnz, row_cols, row_vals,
                                      col_active, accum);
  k_dead<<<DICT / 256, 256, 0, stream>>>(col_active, nba, dead, nba_d, ctl);
  k_decode<<<BATCH, 256, 0, stream>>>(row_nnz, row_cols, row_vals, W_dec, b_dec, y_proc,
                                      y_mean, y_std, y_pred, y_out_d, accum);
  k_auxsel<<<BATCH, 256, 0, stream>>>(acts_dense, dead, aux_nnz, aux_cols, aux_vals);
  k_auxdec<<<BATCH, 256, 0, stream>>>(aux_nnz, aux_cols, aux_vals, W_dec, y_proc, y_pred, accum);
  k_final<<<1, 1, 0, stream>>>(accum, ctl, scalars_d);
}
